// SelfAttention_5935644803226
// MI455X (gfx1250) — matrix operand order repaired and checked by reading, not compiled, not run on hardware
//
#include <hip/hip_runtime.h>
#include <hip/hip_bf16.h>

// ---------------------------------------------------------------------------
// Self-attention block (AttnBlock style) for MI455X / gfx1250, wave32 WMMA.
//   B=4, C=128, H=W=64 -> N=4096 pixels, 4 heads, head_dim=32.
// All GEMMs run on v_wmma_f32_16x16x32_f16 with fp32 accumulation.
// Attention computes S^T = K*Q^T so softmax is lane-local and the P tile
// converts to an A-fragment without any cross-lane movement. K/V chunks are
// staged through double-buffered LDS, using CDNA5 async global->LDS DMA
// (ASYNCcnt) when the toolchain exposes the builtins.
// ---------------------------------------------------------------------------

typedef __attribute__((ext_vector_type(16))) _Float16 v16h;
typedef __attribute__((ext_vector_type(8)))  float    v8f;

#define CCH   128          // channels
#define NPIX  4096         // H*W
#define NH    4            // heads
#define HD    32           // head dim

// --- CDNA5 async global->LDS path (guarded; falls back to uint4 copies) ----
#if defined(__gfx1250__) && __has_builtin(__builtin_amdgcn_global_load_async_to_lds_b128)
#define ASYNC_LDS 1
// Builtin expects AS(1)/AS(3) pointers to a GCC-vector int4 (clang prints
// AS(1) as "__device__" in HIP mode; pointee is int __vector(4)).
typedef int v4i __attribute__((vector_size(16)));
typedef __attribute__((address_space(1))) v4i g_v4i;
typedef __attribute__((address_space(3))) v4i l_v4i;
#else
#define ASYNC_LDS 0
#endif

__device__ __forceinline__ void stage16(const _Float16* gsrc, _Float16* lds) {
#if ASYNC_LDS
    __builtin_amdgcn_global_load_async_to_lds_b128((g_v4i*)gsrc, (l_v4i*)lds, 0, 0);
#else
    *(uint4*)lds = *(const uint4*)gsrc;
#endif
}

__device__ __forceinline__ void async_wait0() {
#if ASYNC_LDS
#if __has_builtin(__builtin_amdgcn_s_wait_asynccnt)
    __builtin_amdgcn_s_wait_asynccnt(0);
#else
    asm volatile("s_wait_asynccnt 0x0" ::: "memory");
#endif
#endif
}

// A-fragment (16x32 f16, M x K): lane l16 = row M; element e -> K index:
//   lanes 0-15 : K in {0..15},  lanes 16-31 : K in {16..31}
__device__ __forceinline__ int a_k(int e, int half) {
    return (half << 4) + e;
}

// WMMA wrapper: D = A*B + C  (f16 in, f32 accumulate)
__device__ __forceinline__ v8f wmma_f16(v16h a, v16h b, v8f c) {
    return __builtin_amdgcn_wmma_f32_16x16x32_f16(false, a, false, b, (short)0, c,
                                                  false, false);
}

// ---------------------------------------------------------------------------
// Kernel 1: fused QKV projection.
//   y[b][p][o] = sum_c x[b][c][p] * W[o][c] + bias[o], stored f16 pixel-major.
// Each wave handles a 16-pixel tile: shared A fragments (X^T), 96 WMMAs.
// ---------------------------------------------------------------------------
__global__ void qkv_proj_kernel(const float* __restrict__ x,
                                const float* __restrict__ wq, const float* __restrict__ bq,
                                const float* __restrict__ wk, const float* __restrict__ bk,
                                const float* __restrict__ wv, const float* __restrict__ bv,
                                _Float16* __restrict__ qF,
                                _Float16* __restrict__ kF,
                                _Float16* __restrict__ vF) {
    const int wave = threadIdx.x >> 5;
    const int lane = threadIdx.x & 31;
    const int half = lane >> 4;
    const int l16  = lane & 15;
    const int tile = blockIdx.x * 4 + wave;   // 0..1023 pixel tiles
    const int b    = tile >> 8;               // 256 tiles per batch
    const int p0   = (tile & 255) << 4;

    // A fragments of X^T: 16 pixels x 32 channels, 4 K-chunks (C=128).
    v16h aX[4];
#pragma unroll
    for (int kc = 0; kc < 4; ++kc) {
#pragma unroll
        for (int e = 0; e < 16; ++e) {
            const int c = kc * 32 + a_k(e, half);
            aX[kc][e] = (_Float16)x[((size_t)b * CCH + c) * NPIX + p0 + l16];
        }
    }

    const float*    Ws[3] = {wq, wk, wv};
    const float*    Bs[3] = {bq, bk, bv};
    _Float16*       Os[3] = {qF, kF, vF};

#pragma unroll
    for (int m = 0; m < 3; ++m) {
        const float* W = Ws[m];
        _Float16*    O = Os[m];
#pragma unroll
        for (int ot = 0; ot < 8; ++ot) {
            const int o = ot * 16 + l16;      // B-fragment column = output channel
            v8f acc = {};
#pragma unroll
            for (int kc = 0; kc < 4; ++kc) {
                v16h bW;                      // B = W^T tile: K=c contiguous per lane
#pragma unroll
                for (int e = 0; e < 16; ++e) {
                    const int c = kc * 32 + half * 16 + e;
                    bW[e] = (_Float16)W[(size_t)o * CCH + c];
                }
                acc = wmma_f16(aX[kc], bW, acc);
            }
            const float bias = Bs[m][o];
#pragma unroll
            for (int r = 0; r < 8; ++r) {     // D: row p0+r+half*8, col o
                const int p = p0 + r + half * 8;
                O[((size_t)b * NPIX + p) * CCH + ot * 16 + l16] =
                    (_Float16)(acc[r] + bias);
            }
        }
    }
}

// ---------------------------------------------------------------------------
// Kernel 2: flash attention, transposed score tiles, double-buffered async
// K/V staging. One wave = one 16-row n-tile; block (4 waves) shares one (b,h).
//   S^T[m][n] = sum_d K[m][d] * Q[n][d]  -> A = K tile, B = Q tile.
// Pipeline per 32-m chunk: wait_async -> barrier -> issue DMA for next chunk
// into the other buffer -> 4 WMMAs + lane-local online softmax.
// ---------------------------------------------------------------------------
__global__ void attn_kernel(const _Float16* __restrict__ qF,
                            const _Float16* __restrict__ kF,
                            const _Float16* __restrict__ vF,
                            _Float16* __restrict__ aoF) {
    __shared__ __align__(16) _Float16 kTile[2][32 * 32];   // [buf][m][d]
    __shared__ __align__(16) _Float16 vTile[2][32 * 32];   // [buf][m][d]

    const int wave = threadIdx.x >> 5;
    const int lane = threadIdx.x & 31;
    const int half = lane >> 4;
    const int l16  = lane & 15;

    const int bh = blockIdx.x >> 6;                     // 0..15
    const int b  = bh >> 2;
    const int h  = bh & 3;
    const int n0 = ((blockIdx.x & 63) << 6) + (wave << 4);

    const size_t base = ((size_t)b * NPIX) * CCH + (size_t)h * HD;

    // Per-thread staging slot: 128 threads x 16B cover one 32x32 f16 tile.
    const int mrow   = threadIdx.x >> 2;
    const int seg    = (threadIdx.x & 3) * 8;
    const int ldsOff = mrow * 32 + seg;

    // Q as loop-invariant B-fragment: column n = l16, K = d = half*16 + e
    v16h bQ;
#pragma unroll
    for (int e = 0; e < 16; ++e)
        bQ[e] = qF[base + (size_t)(n0 + l16) * CCH + half * 16 + e];

    v8f o0 = {}, o1 = {};            // O rows n (elem r -> n0+r+half*8), cols d
    float rm = -1e30f, rs = 0.f;     // softmax stats for row n = n0 + l16

    // Preload chunk 0 into buffer 0.
    {
        const size_t g = base + (size_t)mrow * CCH + seg;
        stage16(&kF[g], &kTile[0][ldsOff]);
        stage16(&vF[g], &vTile[0][ldsOff]);
    }

    for (int m0 = 0; m0 < NPIX; m0 += 32) {
        const int cur = (m0 >> 5) & 1;

        async_wait0();        // my DMA (incl. current buffer) has written LDS
        __syncthreads();      // publish to all waves; everyone done with prev

        // Kick off DMA for the next chunk into the other buffer; it overlaps
        // the WMMAs + softmax below. (Reads of that buffer's old contents
        // completed before each wave's last WMMA of the previous iteration.)
        if (m0 + 32 < NPIX) {
            const size_t g = base + (size_t)(m0 + 32 + mrow) * CCH + seg;
            stage16(&kF[g], &kTile[cur ^ 1][ldsOff]);
            stage16(&vF[g], &vTile[cur ^ 1][ldsOff]);
#if !ASYNC_LDS
            __builtin_prefetch(&kF[g + (size_t)32 * CCH], 0, 1);
            __builtin_prefetch(&vF[g + (size_t)32 * CCH], 0, 1);
#endif
        }

        // S^T tiles: two 16-row m tiles; A = K rows from LDS (2x16B per lane)
        v8f sT[2];
#pragma unroll
        for (int mc = 0; mc < 2; ++mc) {
            v16h aK;
#pragma unroll
            for (int e = 0; e < 16; ++e)
                aK[e] = kTile[cur][(mc * 16 + l16) * 32 + a_k(e, half)];
            v8f z = {};
            sT[mc] = wmma_f16(aK, bQ, z);
        }

        // Lane-local online softmax for row n = n0 + l16.
        float tmax = sT[0][0];
#pragma unroll
        for (int r = 1; r < 8; ++r) tmax = fmaxf(tmax, sT[0][r]);
#pragma unroll
        for (int r = 0; r < 8; ++r) tmax = fmaxf(tmax, sT[1][r]);
        tmax = fmaxf(tmax, __shfl_xor(tmax, 16, 32));   // merge lane halves
        const float nmax  = fmaxf(rm, tmax);
        const float scale = __expf(rm - nmax);
        float p0[8], p1[8], psum = 0.f;
#pragma unroll
        for (int r = 0; r < 8; ++r) {
            p0[r] = __expf(sT[0][r] - nmax);
            p1[r] = __expf(sT[1][r] - nmax);
            psum += p0[r] + p1[r];
        }
        psum += __shfl_xor(psum, 16, 32);
        rs = rs * scale + psum;
        rm = nmax;

        // Rescale O: element r is row n0+r+half*8; its scale lives in lane
        // (r + half*8) -> one bpermute per element row.
#pragma unroll
        for (int r = 0; r < 8; ++r) {
            const float sc = __shfl(scale, r + half * 8, 32);
            o0[r] *= sc;
            o1[r] *= sc;
        }

        // A = P fragment straight from registers: element e has K(m_local) =
        // (e<8?0:16)+half*8+(e&7), exactly sT[0][e] / sT[1][e-8] positions.
        v16h aP;
#pragma unroll
        for (int e = 0; e < 8; ++e) {
            aP[e]     = (_Float16)p0[e];
            aP[e + 8] = (_Float16)p1[e];
        }

        // B = V^T (K = m rows, N = d columns) from LDS.
        v16h bV0, bV1;
#pragma unroll
        for (int e = 0; e < 16; ++e) {
            const int m = ((e & 8) << 1) + (half << 3) + (e & 7);
            bV0[e] = vTile[cur][m * 32 + l16];
            bV1[e] = vTile[cur][m * 32 + 16 + l16];
        }
        o0 = wmma_f16(aP, bV0, o0);
        o1 = wmma_f16(aP, bV1, o1);
    }

    // Normalize (inverse row-sum lives in lane n&15) and emit f16 output.
    const float inv = 1.0f / rs;
#pragma unroll
    for (int r = 0; r < 8; ++r) {
        const float invr = __shfl(inv, r + half * 8, 32);
        const int   n    = n0 + r + half * 8;
        const size_t o   = base + (size_t)n * CCH;
        aoF[o + l16]      = (_Float16)(o0[r] * invr);
        aoF[o + 16 + l16] = (_Float16)(o1[r] * invr);
    }
}

// ---------------------------------------------------------------------------
// Kernel 3: output projection + residual, back to NCHW fp32.
//   out[b][c][p] = gamma * (sum_c' Wo[c][c'] * ao[b][p][c'] + bo[c]) + x[b][c][p]
// ---------------------------------------------------------------------------
__global__ void out_proj_kernel(const _Float16* __restrict__ aoF,
                                const float* __restrict__ wo,
                                const float* __restrict__ bo,
                                const float* __restrict__ gamma,
                                const float* __restrict__ x,
                                float* __restrict__ out) {
    const int wave = threadIdx.x >> 5;
    const int lane = threadIdx.x & 31;
    const int half = lane >> 4;
    const int l16  = lane & 15;

    const int tile = blockIdx.x * 4 + wave;  // 0..8191
    const int b    = tile >> 11;             // 2048 tiles per batch
    const int rem  = tile & 2047;
    const int c0   = (rem >> 8) << 4;        // output-channel tile
    const int p0   = (rem & 255) << 4;       // pixel tile

    v8f acc = {};
#pragma unroll
    for (int kc = 0; kc < 4; ++kc) {
        v16h aW, bA;
#pragma unroll
        for (int e = 0; e < 16; ++e) {
            aW[e] = (_Float16)wo[(size_t)(c0 + l16) * CCH + kc * 32 + a_k(e, half)];
            bA[e] = aoF[((size_t)b * NPIX + p0 + l16) * CCH + kc * 32 + half * 16 + e];
        }
        acc = wmma_f16(aW, bA, acc);
    }

    const float g = gamma[0];
#pragma unroll
    for (int r = 0; r < 8; ++r) {
        const int c = c0 + r + half * 8;
        const size_t idx = ((size_t)b * CCH + c) * NPIX + p0 + l16;
        out[idx] = g * (acc[r] + bo[c]) + x[idx];
    }
}

// ---------------------------------------------------------------------------
extern "C" void kernel_launch(void* const* d_in, const int* in_sizes, int n_in,
                              void* d_out, int out_size, void* d_ws, size_t ws_size,
                              hipStream_t stream) {
    const float* x     = (const float*)d_in[0];
    const float* wq    = (const float*)d_in[1];
    const float* bq    = (const float*)d_in[2];
    const float* wk    = (const float*)d_in[3];
    const float* bk    = (const float*)d_in[4];
    const float* wv    = (const float*)d_in[5];
    const float* bv    = (const float*)d_in[6];
    const float* wo    = (const float*)d_in[7];
    const float* bo    = (const float*)d_in[8];
    const float* gamma = (const float*)d_in[9];
    float* out = (float*)d_out;

    // Workspace: Q, K, V, AO in f16 pixel-major, 4 MB each (16 MB total).
    const size_t fElems = (size_t)4 * NPIX * CCH;
    _Float16* qF  = (_Float16*)d_ws;
    _Float16* kF  = qF + fElems;
    _Float16* vF  = kF + fElems;
    _Float16* aoF = vF + fElems;

    qkv_proj_kernel<<<256, 128, 0, stream>>>(x, wq, bq, wk, bk, wv, bv, qF, kF, vF);
    attn_kernel<<<1024, 128, 0, stream>>>(qF, kF, vF, aoF);
    out_proj_kernel<<<2048, 128, 0, stream>>>(aoF, wo, bo, gamma, x, out);
}
